// IdpSAM_AttentionBlock_20985210208305
// MI455X (gfx1250) — matrix operand order repaired and checked by reading, not compiled, not run on hardware
//
#include <hip/hip_runtime.h>
#include <hip/hip_bf16.h>

// ---------------------------------------------------------------------------
// IdpSAM attention block for gfx1250 (MI455X).
// All GEMMs run on v_wmma_f32_16x16x32_f16 with double-buffered LDS tiles.
// B=4, L=512, D=1024, ND=512, ED=64, H=16, MLP=2048, DH=64.
// ---------------------------------------------------------------------------

typedef __attribute__((ext_vector_type(16))) _Float16 v16h;
typedef __attribute__((ext_vector_type(8)))  _Float16 v8h;
typedef __attribute__((ext_vector_type(8)))  float    v8f;

static constexpr int Bc   = 4;
static constexpr int Lc   = 512;
static constexpr int Dc   = 1024;
static constexpr int NDc  = 512;
static constexpr int Hc   = 16;
static constexpr int MLPc = 2048;
static constexpr int DHc  = 64;
static constexpr int BLc  = Bc * Lc;     // 2048
static constexpr int SIXD = 6 * Dc;      // 6144

// ------------------------------ utility kernels ----------------------------

__global__ void cvt_kernel(const float* __restrict__ src, _Float16* __restrict__ dst, long n) {
  long i = (long)blockIdx.x * blockDim.x + threadIdx.x;
  long stride = (long)gridDim.x * blockDim.x;
  for (; i < n; i += stride) dst[i] = (_Float16)src[i];
}

__global__ void zero_kernel(float* __restrict__ p, int n) {
  int i = blockIdx.x * blockDim.x + threadIdx.x;
  if (i < n) p[i] = 0.f;
}

// Vt[b,h,d,j] = qkv[(b*L+j), 2D + h*64 + d]
__global__ void vt_kernel(const _Float16* __restrict__ qkv, _Float16* __restrict__ vt, long n) {
  long i = (long)blockIdx.x * blockDim.x + threadIdx.x;
  long stride = (long)gridDim.x * blockDim.x;
  for (; i < n; i += stride) {
    long j = i & 511;
    long d = (i >> 9) & 63;
    long h = (i >> 15) & 15;
    long b = i >> 19;
    vt[i] = qkv[(b * 512 + j) * 3072 + 2048 + h * 64 + d];
  }
}

// LN with gamma/beta -> f16 (adaLN input path)
__global__ void ln_gamma_kernel(const float* __restrict__ x, const float* __restrict__ g,
                                const float* __restrict__ b, _Float16* __restrict__ out, int C) {
  __shared__ float r1[256], r2[256];
  long row = blockIdx.x;
  const float* xr = x + row * (long)C;
  int tid = threadIdx.x;
  float s1 = 0.f, s2 = 0.f;
  for (int c = tid; c < C; c += blockDim.x) { float v = xr[c]; s1 += v; s2 += v * v; }
  r1[tid] = s1; r2[tid] = s2; __syncthreads();
  for (int st = 128; st > 0; st >>= 1) {
    if (tid < st) { r1[tid] += r1[tid + st]; r2[tid] += r2[tid + st]; }
    __syncthreads();
  }
  float mean = r1[0] / C;
  float var  = r2[0] / C - mean * mean;
  float rstd = rsqrtf(var + 1e-5f);
  _Float16* orow = out + row * (long)C;
  for (int c = tid; c < C; c += blockDim.x)
    orow[c] = (_Float16)(((xr[c] - mean) * rstd) * g[c] + b[c]);
}

// out = LN(x) * (1 + mod[:,scOff:]) + mod[:,shOff:]  -> f16
__global__ void ln_mod_kernel(const float* __restrict__ x, const float* __restrict__ mod,
                              int ldm, int shOff, int scOff, _Float16* __restrict__ out, int C) {
  __shared__ float r1[256], r2[256];
  long row = blockIdx.x;
  const float* xr = x + row * (long)C;
  int tid = threadIdx.x;
  float s1 = 0.f, s2 = 0.f;
  for (int c = tid; c < C; c += blockDim.x) { float v = xr[c]; s1 += v; s2 += v * v; }
  r1[tid] = s1; r2[tid] = s2; __syncthreads();
  for (int st = 128; st > 0; st >>= 1) {
    if (tid < st) { r1[tid] += r1[tid + st]; r2[tid] += r2[tid + st]; }
    __syncthreads();
  }
  float mean = r1[0] / C;
  float var  = r2[0] / C - mean * mean;
  float rstd = rsqrtf(var + 1e-5f);
  const float* mrow = mod + row * (long)ldm;
  _Float16* orow = out + row * (long)C;
  for (int c = tid; c < C; c += blockDim.x)
    orow[c] = (_Float16)(((xr[c] - mean) * rstd) * (1.f + mrow[scOff + c]) + mrow[shOff + c]);
}

// Edge bias: bias[b,h,i,j] = sum_e (LN(p[b,i,j,:])*lnw+lnb)[e] * edge_w[h,e] + edge_b[h]
// One wave per (b,i,j); ED = 64 (2 floats / lane).
__global__ void edge_bias_kernel(const float* __restrict__ p,
                                 const float* __restrict__ lnw, const float* __restrict__ lnb,
                                 const float* __restrict__ ew, const float* __restrict__ eb,
                                 float* __restrict__ biasOut) {
  __shared__ float sEw[16 * 64];
  __shared__ float sLnw[64], sLnb[64];
  __shared__ float sY[8][64];
  for (int t = threadIdx.x; t < 1024; t += 256) sEw[t] = ew[t];
  if (threadIdx.x < 64) { sLnw[threadIdx.x] = lnw[threadIdx.x]; sLnb[threadIdx.x] = lnb[threadIdx.x]; }
  __syncthreads();

  int wid = threadIdx.x >> 5, lane = threadIdx.x & 31;
  long idx = (long)blockIdx.x * 8 + wid;   // flat (b*L + i)*L + j
  const float* pr = p + idx * 64;
  float2 v = *(const float2*)(pr + lane * 2);
  float s1 = v.x + v.y, s2 = v.x * v.x + v.y * v.y;
  for (int m = 16; m; m >>= 1) { s1 += __shfl_xor(s1, m, 32); s2 += __shfl_xor(s2, m, 32); }
  float mean = s1 * (1.f / 64.f);
  float var  = s2 * (1.f / 64.f) - mean * mean;
  float rstd = rsqrtf(var + 1e-5f);
  int e0 = lane * 2;
  sY[wid][e0]     = (v.x - mean) * rstd * sLnw[e0]     + sLnb[e0];
  sY[wid][e0 + 1] = (v.y - mean) * rstd * sLnw[e0 + 1] + sLnb[e0 + 1];
  __syncthreads();

  if (lane < 16) {
    const float* wrow = sEw + lane * 64;
    float acc = 0.f;
    #pragma unroll
    for (int e = 0; e < 64; ++e) acc += sY[wid][e] * wrow[e];
    acc += eb[lane];
    long j = idx & 511;
    long bi = idx >> 9;
    long i = bi & 511;
    long b = bi >> 9;
    biasOut[((b * 16 + lane) * 512 + i) * 512 + j] = acc;
  }
}

// Softmax over rows of [rows, C] f32 -> f16
__global__ void softmax_kernel(const float* __restrict__ S, _Float16* __restrict__ P, int C) {
  __shared__ float red[256];
  long row = blockIdx.x;
  const float* sr = S + row * (long)C;
  _Float16* pr = P + row * (long)C;
  int tid = threadIdx.x;
  float m = -1e30f;
  for (int c = tid; c < C; c += blockDim.x) m = fmaxf(m, sr[c]);
  red[tid] = m; __syncthreads();
  for (int st = 128; st > 0; st >>= 1) {
    if (tid < st) red[tid] = fmaxf(red[tid], red[tid + st]);
    __syncthreads();
  }
  m = red[0]; __syncthreads();
  float s = 0.f;
  for (int c = tid; c < C; c += blockDim.x) s += __expf(sr[c] - m);
  red[tid] = s; __syncthreads();
  for (int st = 128; st > 0; st >>= 1) {
    if (tid < st) red[tid] += red[tid + st];
    __syncthreads();
  }
  float inv = __builtin_amdgcn_rcpf(red[0]);
  for (int c = tid; c < C; c += blockDim.x) pr[c] = (_Float16)(__expf(sr[c] - m) * inv);
}

// ------------------------------ WMMA GEMM ----------------------------------
// C[M,N] = A[M,K] @ W[N,K]^T (+ epilogue).  Block tile 128x64, 8 waves (4x2),
// each wave computes 2x2 of 16x16 WMMA tiles.  K stepped by 32, double-
// buffered LDS: one barrier per K-step, global prefetch overlaps WMMA.
// Epilogues:
//   0: f32 store       out = acc + bias[n]
//   1: f16 store       out = acc + bias[n]
//   2: f16 SiLU store  out = silu(acc + bias[n])
//   3: f32 scores      out = acc*alpha + cin[off]     (cin laid out like C)
//   4: f32 res+gate    out = resid + (acc+bias)*mod[:,modOff+n]
// Batch indexing via blockIdx.z: offsets = (z/H)*outer + (z%H)*inner.
template <int EPI>
__global__ __launch_bounds__(256) void gemm_kernel(
    const _Float16* __restrict__ A, long aOuter, long aInner, int lda,
    const _Float16* __restrict__ W, long wOuter, long wInner, int ldw,
    void* __restrict__ Cv, long cOuter, long cInner, int ldc,
    const float* __restrict__ bias,
    const float* __restrict__ cin,
    const float* __restrict__ resid, int ldr,
    const float* __restrict__ mod, int ldm, int modOff,
    float alpha, int K, int Hdiv) {
  __shared__ _Float16 sA[2][128 * 40];   // 32 K-halves per row, padded to 40
  __shared__ _Float16 sW[2][64 * 40];

  const int tid  = threadIdx.x;
  const int lane = tid & 31;
  const int wid  = tid >> 5;
  const int wm   = wid & 3;     // 0..3  -> M offset wm*32
  const int wn   = wid >> 2;    // 0..1  -> N offset wn*32
  const int lr    = lane & 15;
  const int khalf = lane >> 4;  // 0 or 1

  const int batch = blockIdx.z;
  const int bo = batch / Hdiv, bi = batch % Hdiv;
  const _Float16* Ab = A + (long)bo * aOuter + (long)bi * aInner;
  const _Float16* Wb = W + (long)bo * wOuter + (long)bi * wInner;
  const long cOff = (long)bo * cOuter + (long)bi * cInner;

  const int m0 = blockIdx.y * 128;
  const int n0 = blockIdx.x * 64;

  const int ar = tid >> 1, ac = (tid & 1) * 16;   // A: 128 rows x 32 halves
  const int wr = tid >> 2, wc = (tid & 3) * 8;    // W: 64 rows x 32 halves
  const _Float16* arow = Ab + (long)(m0 + ar) * lda + ac;   // per-thread slab base
  const _Float16* wrow = Wb + (long)(n0 + wr) * ldw + wc;

  v8f acc[2][2];
  #pragma unroll
  for (int i = 0; i < 2; ++i)
    #pragma unroll
    for (int j = 0; j < 2; ++j)
      #pragma unroll
      for (int v = 0; v < 8; ++v) acc[i][j][v] = 0.f;

  // prologue: stage K-slab 0 into LDS buffer 0
  {
    uint4 pa0 = *(const uint4*)(arow);
    uint4 pa1 = *(const uint4*)(arow + 8);
    uint4 pw0 = *(const uint4*)(wrow);
    *(uint4*)(&sA[0][ar * 40 + ac])     = pa0;
    *(uint4*)(&sA[0][ar * 40 + ac + 8]) = pa1;
    *(uint4*)(&sW[0][wr * 40 + wc])     = pw0;
  }
  __syncthreads();

  int cur = 0;
  for (int k0 = 0; k0 < K; k0 += 32) {
    const bool more = (k0 + 32) < K;   // uniform across the block
    uint4 na0{}, na1{}, nw0{};
    if (more) {
      na0 = *(const uint4*)(arow + k0 + 32);
      na1 = *(const uint4*)(arow + k0 + 40);
      nw0 = *(const uint4*)(wrow + k0 + 32);
    }

    const _Float16* sAc = sA[cur];
    const _Float16* sWc = sW[cur];
    v16h af[2], bf[2];
    #pragma unroll
    for (int mt = 0; mt < 2; ++mt) {
      // A frag (16x32): lane row = lr, K chunks {base..base+7, base+16..base+23}, base = khalf*8
      const _Float16* pa = &sAc[(wm * 32 + mt * 16 + lr) * 40 + khalf * 8];
      v8h lo = *(const v8h*)pa;
      v8h hi = *(const v8h*)(pa + 16);
      #pragma unroll
      for (int e = 0; e < 8; ++e) { af[mt][e] = lo[e]; af[mt][e + 8] = hi[e]; }
    }
    #pragma unroll
    for (int nt = 0; nt < 2; ++nt) {
      // B frag (32x16): lane col = lr, K chunks {base..base+7, base+16..base+23}, base = khalf*8
      const _Float16* pb = &sWc[(wn * 32 + nt * 16 + lr) * 40 + khalf * 8];
      v8h lo = *(const v8h*)pb;
      v8h hi = *(const v8h*)(pb + 16);
      #pragma unroll
      for (int e = 0; e < 8; ++e) { bf[nt][e] = lo[e]; bf[nt][e + 8] = hi[e]; }
    }

    #pragma unroll
    for (int mt = 0; mt < 2; ++mt)
      #pragma unroll
      for (int nt = 0; nt < 2; ++nt)
        acc[mt][nt] = __builtin_amdgcn_wmma_f32_16x16x32_f16(
            false, af[mt], false, bf[nt], (short)0, acc[mt][nt], false, false);

    if (more) {
      const int nxt = cur ^ 1;
      *(uint4*)(&sA[nxt][ar * 40 + ac])     = na0;
      *(uint4*)(&sA[nxt][ar * 40 + ac + 8]) = na1;
      *(uint4*)(&sW[nxt][wr * 40 + wc])     = nw0;
    }
    __syncthreads();
    cur ^= 1;
  }

  // Epilogue: lane holds (m = v + khalf*8, n = lr) within each 16x16 tile.
  #pragma unroll
  for (int mt = 0; mt < 2; ++mt) {
    #pragma unroll
    for (int nt = 0; nt < 2; ++nt) {
      const int gcol = n0 + wn * 32 + nt * 16 + lr;
      float bv = 0.f;
      if constexpr (EPI == 0 || EPI == 1 || EPI == 2 || EPI == 4) bv = bias[gcol];
      #pragma unroll
      for (int v = 0; v < 8; ++v) {
        const int grow = m0 + wm * 32 + mt * 16 + khalf * 8 + v;
        const long off = cOff + (long)grow * ldc + gcol;
        float x = acc[mt][nt][v];
        if constexpr (EPI == 0) {
          ((float*)Cv)[off] = x + bv;
        } else if constexpr (EPI == 1) {
          ((_Float16*)Cv)[off] = (_Float16)(x + bv);
        } else if constexpr (EPI == 2) {
          float t = x + bv;
          ((_Float16*)Cv)[off] = (_Float16)(t * __builtin_amdgcn_rcpf(1.f + __expf(-t)));
        } else if constexpr (EPI == 3) {
          ((float*)Cv)[off] = x * alpha + cin[off];
        } else {
          float t = x + bv;
          ((float*)Cv)[off] = resid[(long)grow * ldr + gcol] +
                              t * mod[(long)grow * ldm + modOff + gcol];
        }
      }
    }
  }
}

// ------------------------------ launcher -----------------------------------

extern "C" void kernel_launch(void* const* d_in, const int* in_sizes, int n_in,
                              void* d_out, int out_size, void* d_ws, size_t ws_size,
                              hipStream_t stream) {
  const float* h_in  = (const float*)d_in[0];
  const float* s_in  = (const float*)d_in[1];
  const float* p_in  = (const float*)d_in[2];
  const float* alnw  = (const float*)d_in[3];
  const float* alnb  = (const float*)d_in[4];
  const float* aw1   = (const float*)d_in[5];
  const float* ab1   = (const float*)d_in[6];
  const float* aw2   = (const float*)d_in[7];
  const float* ab2   = (const float*)d_in[8];
  const float* elnw  = (const float*)d_in[9];
  const float* elnb  = (const float*)d_in[10];
  const float* e_w   = (const float*)d_in[11];
  const float* e_b   = (const float*)d_in[12];
  const float* in_w  = (const float*)d_in[13];
  const float* in_b  = (const float*)d_in[14];
  const float* out_w = (const float*)d_in[15];
  const float* out_b = (const float*)d_in[16];
  const float* mw1   = (const float*)d_in[17];
  const float* mb1   = (const float*)d_in[18];
  const float* mw2   = (const float*)d_in[19];
  const float* mb2   = (const float*)d_in[20];

  char* base = (char*)d_ws;
  size_t off = 0;
  auto alloc = [&](size_t bytes) -> void* {
    void* r = base + off;
    off = (off + bytes + 255) & ~(size_t)255;
    return r;
  };

  _Float16* in_wh  = (_Float16*)alloc((size_t)3 * Dc * Dc * 2);
  _Float16* out_wh = (_Float16*)alloc((size_t)Dc * Dc * 2);
  _Float16* mw1h   = (_Float16*)alloc((size_t)MLPc * Dc * 2);
  _Float16* mw2h   = (_Float16*)alloc((size_t)Dc * MLPc * 2);
  _Float16* aw1h   = (_Float16*)alloc((size_t)NDc * NDc * 2);
  _Float16* aw2h   = (_Float16*)alloc((size_t)SIXD * NDc * 2);
  _Float16* xs     = (_Float16*)alloc((size_t)BLc * NDc * 2);
  _Float16* t1     = (_Float16*)alloc((size_t)BLc * NDc * 2);
  float*    modb   = (float*)   alloc((size_t)BLc * SIXD * 4);
  float*    biasb  = (float*)   alloc((size_t)Bc * Hc * Lc * Lc * 4);
  _Float16* xattn  = (_Float16*)alloc((size_t)BLc * Dc * 2);
  _Float16* qkvh   = (_Float16*)alloc((size_t)BLc * 3 * Dc * 2);
  _Float16* Pm     = (_Float16*)alloc((size_t)Bc * Hc * Lc * Lc * 2);
  _Float16* Vt     = (_Float16*)alloc((size_t)Bc * Hc * DHc * Lc * 2);
  _Float16* obuf   = (_Float16*)alloc((size_t)BLc * Dc * 2);
  float*    h2     = (float*)   alloc((size_t)BLc * Dc * 4);
  _Float16* xmlp   = (_Float16*)alloc((size_t)BLc * Dc * 2);
  _Float16* m1     = (_Float16*)alloc((size_t)BLc * MLPc * 2);
  float*    zbias  = (float*)   alloc((size_t)256 * 4);
  (void)ws_size; (void)in_sizes; (void)n_in; (void)out_size;

  // --- weight f32 -> f16 conversions + zero bias
  cvt_kernel<<<2048, 256, 0, stream>>>(in_w,  in_wh,  (long)3 * Dc * Dc);
  cvt_kernel<<<2048, 256, 0, stream>>>(out_w, out_wh, (long)Dc * Dc);
  cvt_kernel<<<2048, 256, 0, stream>>>(mw1,   mw1h,   (long)MLPc * Dc);
  cvt_kernel<<<2048, 256, 0, stream>>>(mw2,   mw2h,   (long)Dc * MLPc);
  cvt_kernel<<<2048, 256, 0, stream>>>(aw1,   aw1h,   (long)NDc * NDc);
  cvt_kernel<<<2048, 256, 0, stream>>>(aw2,   aw2h,   (long)SIXD * NDc);
  zero_kernel<<<1, 256, 0, stream>>>(zbias, 256);

  // --- adaLN modulation path
  ln_gamma_kernel<<<BLc, 256, 0, stream>>>(s_in, alnw, alnb, xs, NDc);
  {
    dim3 g(NDc / 64, BLc / 128, 1);
    gemm_kernel<2><<<g, 256, 0, stream>>>(xs, 0, 0, NDc, aw1h, 0, 0, NDc,
        t1, 0, 0, NDc, ab1, nullptr, nullptr, 0, nullptr, 0, 0, 1.f, NDc, 1);
  }
  {
    dim3 g(SIXD / 64, BLc / 128, 1);
    gemm_kernel<0><<<g, 256, 0, stream>>>(t1, 0, 0, NDc, aw2h, 0, 0, NDc,
        modb, 0, 0, SIXD, ab2, nullptr, nullptr, 0, nullptr, 0, 0, 1.f, NDc, 1);
  }

  // --- edge-derived attention bias [B,H,L,L]
  edge_bias_kernel<<<(Bc * Lc * Lc) / 8, 256, 0, stream>>>(p_in, elnw, elnb, e_w, e_b, biasb);

  // --- attention branch
  ln_mod_kernel<<<BLc, 256, 0, stream>>>(h_in, modb, SIXD, 0, Dc, xattn, Dc);
  {
    dim3 g((3 * Dc) / 64, BLc / 128, 1);
    gemm_kernel<1><<<g, 256, 0, stream>>>(xattn, 0, 0, Dc, in_wh, 0, 0, Dc,
        qkvh, 0, 0, 3 * Dc, in_b, nullptr, nullptr, 0, nullptr, 0, 0, 1.f, Dc, 1);
  }
  {
    // scores: per (b,h)  S = Q K^T / 8 + bias   (accumulated into biasb)
    dim3 g(Lc / 64, Lc / 128, Bc * Hc);
    gemm_kernel<3><<<g, 256, 0, stream>>>(
        qkvh,       (long)Lc * 3 * Dc, 64, 3 * Dc,       // Q slice
        qkvh + Dc,  (long)Lc * 3 * Dc, 64, 3 * Dc,       // K slice
        biasb,      (long)Hc * Lc * Lc, (long)Lc * Lc, Lc,
        nullptr, biasb, nullptr, 0, nullptr, 0, 0, 0.125f, DHc, Hc);
  }
  softmax_kernel<<<Bc * Hc * Lc, 256, 0, stream>>>(biasb, Pm, Lc);
  vt_kernel<<<2048, 256, 0, stream>>>(qkvh, Vt, (long)Bc * Hc * DHc * Lc);
  {
    // O = P @ V  per (b,h) -> obuf [B*L, D] with head-interleaved columns
    dim3 g(DHc / 64, Lc / 128, Bc * Hc);
    gemm_kernel<1><<<g, 256, 0, stream>>>(
        Pm, (long)Hc * Lc * Lc, (long)Lc * Lc, Lc,
        Vt, (long)Hc * DHc * Lc, (long)DHc * Lc, Lc,
        obuf, (long)Lc * Dc, DHc, Dc,
        zbias, nullptr, nullptr, 0, nullptr, 0, 0, 1.f, Lc, Hc);
  }
  {
    // h2 = h + (obuf @ out_w^T + out_b) * gate_msa
    dim3 g(Dc / 64, BLc / 128, 1);
    gemm_kernel<4><<<g, 256, 0, stream>>>(obuf, 0, 0, Dc, out_wh, 0, 0, Dc,
        h2, 0, 0, Dc, out_b, nullptr, h_in, Dc, modb, SIXD, 2 * Dc, 1.f, Dc, 1);
  }

  // --- gated MLP branch
  ln_mod_kernel<<<BLc, 256, 0, stream>>>(h2, modb, SIXD, 3 * Dc, 4 * Dc, xmlp, Dc);
  {
    dim3 g(MLPc / 64, BLc / 128, 1);
    gemm_kernel<2><<<g, 256, 0, stream>>>(xmlp, 0, 0, Dc, mw1h, 0, 0, Dc,
        m1, 0, 0, MLPc, mb1, nullptr, nullptr, 0, nullptr, 0, 0, 1.f, Dc, 1);
  }
  {
    // out = h2 + (m1 @ mlp_w2^T + mlp_b2) * gate_mlp
    dim3 g(Dc / 64, BLc / 128, 1);
    gemm_kernel<4><<<g, 256, 0, stream>>>(m1, 0, 0, MLPc, mw2h, 0, 0, MLPc,
        (float*)d_out, 0, 0, Dc, mb2, nullptr, h2, Dc, modb, SIXD, 5 * Dc, 1.f, MLPc, 1);
  }
}
